// LearnableFeaturePermute_13262859010401
// MI455X (gfx1250) — matrix operand order repaired and checked by reading, not compiled, not run on hardware
//
#include <hip/hip_runtime.h>
#include <hip/hip_bf16.h>
#include <math.h>

// ---------------------------------------------------------------------------
// LearnableFeaturePermute on MI455X (gfx1250)
//   Phase 1: Sinkhorn on 512x512 logits (tiny, L2-resident, plain kernels)
//   Phase 2: out[M,N] = x[M,K] * exp(logP)[K,N], M=65536, N=K=512
//     - bf16x3 split-precision WMMA GEMM (hi*hi + hi*lo + lo*hi, f32 acc)
//     - whole B-slice resident in CDNA5's 320KB LDS (hi+lo, transposed)
//     - A tiles DMA'd by the Tensor Data Mover (tensor_load_to_lds) with
//       descriptor-driven LDS row padding; double-buffered via TENSORcnt
// ---------------------------------------------------------------------------

#define FDIM   512
#define TILE_M 128
#define TILE_N 128
#define KSTEP  32
#define NKT    (FDIM / KSTEP)   // 16 k-steps
#define PAF    36               // A LDS pitch in f32 (32 + 4 pad dwords = 144B rows, 16B aligned)
#define PB     528              // B LDS pitch in bf16 (512 + 16 pad = 1056B rows, 16B aligned)

typedef __attribute__((ext_vector_type(16))) __bf16       v16bf;
typedef __attribute__((ext_vector_type(8)))  __bf16       v8bf;
typedef __attribute__((ext_vector_type(8)))  float        v8f;
typedef __attribute__((ext_vector_type(4)))  float        v4f;
typedef __attribute__((ext_vector_type(4)))  unsigned int u32x4;
typedef __attribute__((ext_vector_type(8)))  int          i32x8;
typedef __attribute__((ext_vector_type(4)))  int          i32x4;

__device__ __forceinline__ v16bf cat8(v8bf lo, v8bf hi) {
    return __builtin_shufflevector(lo, hi, 0,1,2,3,4,5,6,7,8,9,10,11,12,13,14,15);
}

__device__ __forceinline__ v8f wmma_bf16(v16bf a, v16bf b, v8f c) {
    // 8 args: (neg_a, A, neg_b, B, c_mod, C, reuse_a, reuse_b)
    return __builtin_amdgcn_wmma_f32_16x16x32_bf16(false, a, false, b, (short)0, c, false, false);
}

// Tensor Data Mover: DMA a [128 rows x 32 f32] tile (row stride FDIM) from
// global to LDS, padding +4 dwords after every 32 dwords -> 36-dword pitch.
__device__ __forceinline__ void tdm_load_A(const float* gsrc, unsigned lds_addr) {
    unsigned long long ga = (unsigned long long)(const void*)gsrc;
    u32x4 g0 = {
        1u,                                                 // count=1, user desc
        lds_addr,                                           // lds_addr (bytes)
        (unsigned)ga,                                       // global_addr[31:0]
        (unsigned)((ga >> 32) & 0x01FFFFFFu) | (2u << 30)   // global_addr[56:32] | type=2
    };
    i32x8 g1 = {
        (2 << 16) | (1 << 20) | (4 << 22) | (3 << 25),      // data_size=4B, pad_en,
                                                            // pad_interval=32dw, pad_amount=4dw
        (FDIM & 0xFFFF) << 16,                              // tensor_dim0 = 512
        0,                                                  // td0_hi=0 | td1_lo (65536&0xFFFF=0)
        1 | (KSTEP << 16),                                  // td1_hi=1 (65536) | tile_dim0=32
        TILE_M,                                             // tile_dim1=128, tile_dim2=0
        FDIM,                                               // tensor_dim0_stride = 512
        0, 0
    };
    i32x4 gz4 = {0, 0, 0, 0};
    i32x8 gz8 = {0, 0, 0, 0, 0, 0, 0, 0};
    // 6-arg toolchain variant: (g0, g1, g2, g3, extra_group, cpol)
    __builtin_amdgcn_tensor_load_to_lds(g0, g1, gz4, gz4, gz8, 0);
}

// ---------------------------------------------------------------------------
// Sinkhorn kernels (512x512, in log space, in d_ws)
// ---------------------------------------------------------------------------

__global__ void sk_copy_kernel(const float* __restrict__ logits, float* __restrict__ logP) {
    int i = blockIdx.x * 256 + threadIdx.x;           // TEMPERATURE == 1.0
    logP[i] = logits[i];
}

__global__ void sk_row_lse_kernel(const float* __restrict__ logP, float* __restrict__ rlse) {
    __shared__ float sm[256];
    const int r = blockIdx.x, t = threadIdx.x;
    float a = logP[r * FDIM + t];
    float b = logP[r * FDIM + t + 256];
    float m = fmaxf(a, b);
    sm[t] = m; __syncthreads();
    for (int s = 128; s > 0; s >>= 1) { if (t < s) sm[t] = fmaxf(sm[t], sm[t + s]); __syncthreads(); }
    const float M = sm[0]; __syncthreads();
    sm[t] = expf(a - M) + expf(b - M); __syncthreads();
    for (int s = 128; s > 0; s >>= 1) { if (t < s) sm[t] += sm[t + s]; __syncthreads(); }
    if (t == 0) rlse[r] = M + logf(sm[0]);
}

__global__ void sk_col_lse_kernel(const float* __restrict__ logP, const float* __restrict__ rlse,
                                  float* __restrict__ clse) {
    __shared__ float sm[256];
    const int c = blockIdx.x, t = threadIdx.x;
    float a = logP[t * FDIM + c] - rlse[t];
    float b = logP[(t + 256) * FDIM + c] - rlse[t + 256];
    float m = fmaxf(a, b);
    sm[t] = m; __syncthreads();
    for (int s = 128; s > 0; s >>= 1) { if (t < s) sm[t] = fmaxf(sm[t], sm[t + s]); __syncthreads(); }
    const float M = sm[0]; __syncthreads();
    sm[t] = expf(a - M) + expf(b - M); __syncthreads();
    for (int s = 128; s > 0; s >>= 1) { if (t < s) sm[t] += sm[t + s]; __syncthreads(); }
    if (t == 0) clse[c] = M + logf(sm[0]);
}

__global__ void sk_update_kernel(float* __restrict__ logP, const float* __restrict__ rlse,
                                 const float* __restrict__ clse) {
    int i = blockIdx.x * 256 + threadIdx.x;
    logP[i] -= rlse[i >> 9] + clse[i & (FDIM - 1)];
}

// ---------------------------------------------------------------------------
// Main GEMM: TDM-fed, bf16x3 split precision, B fully LDS-resident
// ---------------------------------------------------------------------------

__global__ void __launch_bounds__(256)
permute_gemm_kernel(const float* __restrict__ x, const float* __restrict__ logP,
                    float* __restrict__ out) {
    // 270,336 B for B (hi+lo) + 36,864 B double-buffered A (f32) = 307,200 B
    __shared__ alignas(16) __bf16 sBhi[TILE_N * PB];
    __shared__ alignas(16) __bf16 sBlo[TILE_N * PB];
    __shared__ alignas(16) float  sAf32[2][TILE_M * PAF];

    const int tid  = threadIdx.x;
    const int lane = tid & 31;
    const int wave = tid >> 5;
    const int wm   = wave >> 1;           // 0..3 -> 32-row strip
    const int wn   = wave & 1;            // 0..1 -> 64-col strip
    const int nb   = blockIdx.x & 3;
    const long mb  = blockIdx.x >> 2;
    const long m0  = mb * TILE_M;
    const int  n0  = nb * TILE_N;

    // LDS byte offsets of the two A buffers (generic-pointer low 32 bits = LDS addr)
    const unsigned ldsA0 = (unsigned)(unsigned long long)(const void*)&sAf32[0][0];
    const unsigned ldsA1 = (unsigned)(unsigned long long)(const void*)&sAf32[1][0];

    // Kick off the first A-tile DMA before B staging so it overlaps.
    if (wave == 0) tdm_load_A(x + m0 * FDIM, ldsA0);

    // ---- one-time: stage B slice. Fused exp(logP) + bf16 hi/lo split,
    //      stored transposed [n][k] so B fragments are contiguous per lane.
    for (int idx = tid; idx < FDIM * TILE_N; idx += 256) {
        int k = idx >> 7;                 // 0..511 (coalesced over n)
        int n = idx & 127;                // 0..127
        float p  = expf(logP[k * FDIM + n0 + n]);
        __bf16 h = (__bf16)p;
        __bf16 l = (__bf16)(p - (float)h);
        sBhi[n * PB + k] = h;
        sBlo[n * PB + k] = l;
    }
    if (wave == 0) __builtin_amdgcn_s_wait_tensorcnt(0);
    __syncthreads();

    v8f acc[2][4];
    {
        v8f z = {0.f, 0.f, 0.f, 0.f, 0.f, 0.f, 0.f, 0.f};
        #pragma unroll
        for (int mi = 0; mi < 2; ++mi)
            #pragma unroll
            for (int ni = 0; ni < 4; ++ni) acc[mi][ni] = z;
    }

    const int mrow  = lane & 15;          // A: row within 16, both half-waves
    const int khalf = (lane >> 4) * 8;    // A: lanes<16 hold K{0..7,16..23}, lanes>=16 K{8..15,24..31}
    const int ncol  = lane & 15;          // B: column within 16
    const int kgrp  = (lane >> 4) * 8;    // B: lanes<16 K{0..7,16..23}, lanes>=16 K{8..15,24..31}

    for (int kt = 0; kt < NKT; ++kt) {
        const int cur = kt & 1;
        if (kt + 1 < NKT && wave == 0)    // async DMA of the next A tile
            tdm_load_A(x + m0 * FDIM + (kt + 1) * KSTEP, cur ? ldsA0 : ldsA1);

        // ---- A fragments: read f32 from LDS, split to bf16 hi/lo in regs
        //      (v_cvt_pk_bf16_f32 co-executes with the WMMA/XDL pipe)
        v16bf ah[2], al[2], bh[4], bl[4];
        #pragma unroll
        for (int mi = 0; mi < 2; ++mi) {
            int roff = (wm * 32 + mi * 16 + mrow) * PAF + khalf;
            v4f q0 = *(const v4f*)&sAf32[cur][roff];        // K k0+0..3
            v4f q1 = *(const v4f*)&sAf32[cur][roff + 4];    // K k0+4..7
            v4f q2 = *(const v4f*)&sAf32[cur][roff + 16];   // K k0+16..19
            v4f q3 = *(const v4f*)&sAf32[cur][roff + 20];   // K k0+20..23
            v16bf h, l;
            #pragma unroll
            for (int j = 0; j < 4; ++j) {
                float f0 = q0[j], f1 = q1[j], f2 = q2[j], f3 = q3[j];
                __bf16 b0 = (__bf16)f0, b1 = (__bf16)f1, b2 = (__bf16)f2, b3 = (__bf16)f3;
                h[j] = b0; h[4 + j] = b1; h[8 + j] = b2; h[12 + j] = b3;
                l[j]      = (__bf16)(f0 - (float)b0);
                l[4 + j]  = (__bf16)(f1 - (float)b1);
                l[8 + j]  = (__bf16)(f2 - (float)b2);
                l[12 + j] = (__bf16)(f3 - (float)b3);
            }
            ah[mi] = h; al[mi] = l;
        }
        #pragma unroll
        for (int ni = 0; ni < 4; ++ni) {
            int coff = (wn * 64 + ni * 16 + ncol) * PB + kt * KSTEP + kgrp;
            bh[ni] = cat8(*(const v8bf*)&sBhi[coff], *(const v8bf*)&sBhi[coff + 16]);
            bl[ni] = cat8(*(const v8bf*)&sBlo[coff], *(const v8bf*)&sBlo[coff + 16]);
        }

        #pragma unroll
        for (int mi = 0; mi < 2; ++mi)
            #pragma unroll
            for (int ni = 0; ni < 4; ++ni) {
                acc[mi][ni] = wmma_bf16(ah[mi], bh[ni], acc[mi][ni]);  // hi*hi
                acc[mi][ni] = wmma_bf16(ah[mi], bl[ni], acc[mi][ni]);  // hi*lo
                acc[mi][ni] = wmma_bf16(al[mi], bh[ni], acc[mi][ni]);  // lo*hi
            }

        if (kt + 1 < NKT && wave == 0)    // next tile's DMA must land before the barrier
            __builtin_amdgcn_s_wait_tensorcnt(0);
        __syncthreads();
    }

    // ---- epilogue: C layout = VGPR v -> M = v + 8*(lane>=16), N = lane&15
    const int radd = (lane >> 4) * 8;
    #pragma unroll
    for (int mi = 0; mi < 2; ++mi)
        #pragma unroll
        for (int ni = 0; ni < 4; ++ni) {
            long rbase = m0 + wm * 32 + mi * 16 + radd;
            int  col   = n0 + wn * 64 + ni * 16 + (lane & 15);
            #pragma unroll
            for (int v = 0; v < 8; ++v)
                out[(rbase + v) * FDIM + col] = acc[mi][ni][v];
        }
}

// ---------------------------------------------------------------------------
// Host-side launcher
// ---------------------------------------------------------------------------

extern "C" void kernel_launch(void* const* d_in, const int* in_sizes, int n_in,
                              void* d_out, int out_size, void* d_ws, size_t ws_size,
                              hipStream_t stream) {
    const float* x      = (const float*)d_in[0];   // [16,4096,512] f32
    const float* logits = (const float*)d_in[1];   // [512,512] f32
    float*       out    = (float*)d_out;           // [16,4096,512] f32

    // workspace: logP (1MB) + row_lse (2KB) + col_lse (2KB)
    float* logP = (float*)d_ws;
    float* rlse = logP + FDIM * FDIM;
    float* clse = rlse + FDIM;

    const int elems = FDIM * FDIM;                 // 262144
    const int eblks = elems / 256;                 // 1024

    sk_copy_kernel<<<eblks, 256, 0, stream>>>(logits, logP);
    for (int it = 0; it < 10; ++it) {
        sk_row_lse_kernel<<<FDIM, 256, 0, stream>>>(logP, rlse);
        sk_col_lse_kernel<<<FDIM, 256, 0, stream>>>(logP, rlse, clse);
        sk_update_kernel<<<eblks, 256, 0, stream>>>(logP, rlse, clse);
    }

    const int M = 16 * 4096;                         // 65536
    const int grid = (M / TILE_M) * (FDIM / TILE_N); // 512 * 4 = 2048
    permute_gemm_kernel<<<grid, 256, 0, stream>>>(x, logP, out);
}
